// MHBAMixerV2Block_45724221833614
// MI455X (gfx1250) — matrix operand order repaired and checked by reading, not compiled, not run on hardware
//
#include <hip/hip_runtime.h>

typedef __attribute__((ext_vector_type(16))) _Float16 v16h;
typedef __attribute__((ext_vector_type(8)))  float    v8f;
typedef __attribute__((ext_vector_type(4)))  float    f4;

#define B_    8
#define S_    4096
#define H_    16
#define D_    64
#define HID_  1024
#define ST_   256                 // S_/16
#define NTILE_ (B_ * H_ * ST_)    // 32768 16-token tiles

// Wave-local LDS ordering across the cross-lane transpose handoff.
// DS ops from one wave are in-order at the LDS unit; this pins the compiler.
__device__ __forceinline__ void lds_fence() {
  asm volatile("s_wait_dscnt 0" ::: "memory");
}

__device__ __forceinline__ v8f wmma_f16(v16h a, v16h b, v8f c) {
  // D = A(16x32,f16) * B(32x16,f16) + C(16x16,f32)
  return __builtin_amdgcn_wmma_f32_16x16x32_f16(
      /*neg_a=*/false, a, /*neg_b=*/false, b,
      /*c_mod=*/(short)0, c, /*reuse_a=*/false, /*reuse_b=*/false);
}

// sum across the 16-lane half-groups (N-lanes of a WMMA C tile)
__device__ __forceinline__ float rsum16(float v) {
  v += __shfl_xor(v, 1);
  v += __shfl_xor(v, 2);
  v += __shfl_xor(v, 4);
  v += __shfl_xor(v, 8);
  return v;
}

// B fragment: lane holds column n = nb*16 + (lane&15), K = kb*32 + (lane>=16?8:0) + e (e<8), +16 for e>=8
// Weights stored [N][K] row-major f16 in LDS -> two aligned 128-bit reads.
__device__ __forceinline__ v16h load_bfrag(const _Float16* w, int Kdim, int nb, int kb, int lane) {
  const int n = nb * 16 + (lane & 15);
  const _Float16* p = w + n * Kdim + kb * 32 + ((lane >> 4) << 3);
  typedef __attribute__((ext_vector_type(8))) _Float16 v8h_;
  const v8h_ lo = *reinterpret_cast<const v8h_*>(p);
  const v8h_ hi = *reinterpret_cast<const v8h_*>(p + 16);
  return __builtin_shufflevector(lo, hi, 0,1,2,3,4,5,6,7,8,9,10,11,12,13,14,15);
}

// A fragment straight from GLOBAL memory (streaming, non-temporal).
// rowp = row base + hf*8; elems 0..7 -> K = kbase + hf*8 + e,
// elems 8..15 -> K = kbase + 16 + hf*8 + e  (ISA 16-bit A 16x32 layout)
__device__ __forceinline__ v16h afrag_global(const float* rowp, int kbase) {
  const f4 x0 = __builtin_nontemporal_load((const f4*)(rowp + kbase));
  const f4 x1 = __builtin_nontemporal_load((const f4*)(rowp + kbase + 4));
  const f4 y0 = __builtin_nontemporal_load((const f4*)(rowp + kbase + 16));
  const f4 y1 = __builtin_nontemporal_load((const f4*)(rowp + kbase + 20));
  v16h a;
#pragma unroll
  for (int e = 0; e < 4; ++e) {
    a[e]      = (_Float16)x0[e];
    a[e + 4]  = (_Float16)x1[e];
    a[e + 8]  = (_Float16)y0[e];
    a[e + 12] = (_Float16)y1[e];
  }
  return a;
}

// A fragment from the LDS transpose buffer (row stride in floats).
__device__ __forceinline__ v16h afrag_lds(const float* buf, int stride, int row, int hf, int kbase) {
  const float* p0 = buf + row * stride + kbase + hf * 8;
  const float* p1 = p0 + 16;
  v16h a;
#pragma unroll
  for (int e = 0; e < 8; ++e) {
    a[e]     = (_Float16)p0[e];
    a[e + 8] = (_Float16)p1[e];
  }
  return a;
}

__global__ __launch_bounds__(256) void mixer_kernel(
    const float* __restrict__ gq, const float* __restrict__ gk, const float* __restrict__ gv,
    const float* __restrict__ gmem,
    const float* __restrict__ Wq, const float* __restrict__ Wk, const float* __restrict__ Wv,
    const float* __restrict__ fg,
    const float* __restrict__ ln1w, const float* __restrict__ ln1b,
    const float* __restrict__ W1,
    const float* __restrict__ ln2w, const float* __restrict__ ln2b,
    const float* __restrict__ W2, const float* __restrict__ b2,
    float* __restrict__ out)
{
  extern __shared__ char smem[];
  _Float16* wq = (_Float16*)smem;        // [64][64]   8 KB
  _Float16* wk = wq + 64 * 64;           // [64][64]   8 KB
  _Float16* wv = wk + 64 * 64;           // [64][64]   8 KB
  _Float16* w1 = wv + 64 * 64;           // [128][64] 16 KB
  _Float16* w2 = w1 + 128 * 64;          // [64][128] 16 KB
  const int tid  = threadIdx.x;
  const int warp = tid >> 5;
  const int lane = tid & 31;
  float* buf = (float*)(smem + 57344) + warp * 2112;   // 8448 B per-wave LN scratch

  // --- block-cooperative f32->f16 weight staging (hot in LDS for all tiles) ---
  for (int i = tid; i < 64 * 64; i += 256) {
    wq[i] = (_Float16)Wq[i]; wk[i] = (_Float16)Wk[i]; wv[i] = (_Float16)Wv[i];
  }
  for (int i = tid; i < 128 * 64; i += 256) {
    w1[i] = (_Float16)W1[i]; w2[i] = (_Float16)W2[i];
  }
  __syncthreads();

  const int nl = lane & 15;   // N-lane within C tile / A-row index
  const int hf = lane >> 4;   // half select

  // per-lane constants (depend only on N = nb*16+nl), hoisted out of the tile loop
  float fgl[4], l1wv[4], l1bv[4], b2v[4];
#pragma unroll
  for (int nb = 0; nb < 4; ++nb) {
    const int N = nb * 16 + nl;
    fgl[nb] = fg[N]; l1wv[nb] = ln1w[N]; l1bv[nb] = ln1b[N]; b2v[nb] = b2[N];
  }
  float l2wv[8], l2bv[8];
#pragma unroll
  for (int nb = 0; nb < 8; ++nb) {
    const int N = nb * 16 + nl;
    l2wv[nb] = ln2w[N]; l2bv[nb] = ln2b[N];
  }

  const size_t OUT2 = (size_t)B_ * S_ * HID_;   // cur_mem follows `out` in d_out
  const size_t arow = (size_t)nl * HID_ + hf * 8;  // per-lane A-row offset in q/k/v
  const int wg = blockIdx.x * 8 + warp;
  const int ws = gridDim.x * 8;

  for (int t = wg; t < NTILE_; t += ws) {
    const int b   = t / (H_ * ST_);
    const int rem = t - b * (H_ * ST_);
    const int h   = rem / ST_;
    const int st  = rem - h * ST_;
    const int s0  = st * 16;
    const size_t aoff = ((size_t)b * S_ + s0) * HID_ + (size_t)h * D_;  // [b,s,h*64]
    const size_t moff = (((size_t)b * H_ + h) * S_ + s0) * D_;          // [b,h,s,0]
    const float* mp = gmem + moff;
    float* o1 = out + aoff;
    float* o2 = out + OUT2 + moff;

    // prefetch next tile through the gfx1250 prefetch path
    if (t + ws < NTILE_) {
      const int t2   = t + ws;
      const int pb   = t2 / (H_ * ST_);
      const int rem2 = t2 - pb * (H_ * ST_);
      const int ph   = rem2 / ST_;
      const int pst  = rem2 - ph * ST_;
      const size_t na = ((size_t)pb * S_ + pst * 16) * HID_ + (size_t)ph * D_;
      const size_t nm = (((size_t)pb * H_ + ph) * S_ + pst * 16) * D_;
      const size_t po = (size_t)nl * HID_ + (size_t)hf * 32;
      __builtin_prefetch(gq + na + po, 0, 0);
      __builtin_prefetch(gk + na + po, 0, 0);
      __builtin_prefetch(gv + na + po, 0, 0);
      __builtin_prefetch(gmem + nm + (size_t)lane * 32, 0, 0);
    }

    // ---- f16 A-fragments built directly from global (NT streaming loads) ----
    const float* qr = gq + aoff + arow;
    const float* kr = gk + aoff + arow;
    const float* vr = gv + aoff + arow;
    v16h aq0 = afrag_global(qr, 0), aq1 = afrag_global(qr, 32);
    v16h ak0 = afrag_global(kr, 0), ak1 = afrag_global(kr, 32);
    v16h av0 = afrag_global(vr, 0), av1 = afrag_global(vr, 32);

    // ---- q/k/v projections (WMMA) + gated memory cell + cv, fused per N-block ----
    v8f cvv[4];
#pragma unroll
    for (int nb = 0; nb < 4; ++nb) {
      v8f ka = {};
      ka = wmma_f16(ak0, load_bfrag(wk, 64, nb, 0, lane), ka);
      ka = wmma_f16(ak1, load_bfrag(wk, 64, nb, 1, lane), ka);
      v8f va = {};
      va = wmma_f16(av0, load_bfrag(wv, 64, nb, 0, lane), va);
      va = wmma_f16(av1, load_bfrag(wv, 64, nb, 1, lane), va);
      v8f qa = {};
      qa = wmma_f16(aq0, load_bfrag(wq, 64, nb, 0, lane), qa);
      qa = wmma_f16(aq1, load_bfrag(wq, 64, nb, 1, lane), qa);

      const int   N = nb * 16 + nl;
      const float f = fgl[nb];
      v8f cv_;
#pragma unroll
      for (int r = 0; r < 8; ++r) {
        const int   M    = r + hf * 8;                 // token row in tile
        const float m    = __builtin_nontemporal_load(mp + (size_t)M * D_ + N);
        const float cell = ka[r] * va[r] + f * m;
        const float cm   = (1.f - f) * cell + f * m;
        __builtin_nontemporal_store(cm, o2 + (size_t)M * D_ + N);   // cur_mem
        cv_[r] = qa[r] * cm;
      }
      cvv[nb] = cv_;
    }

    // ---- LayerNorm 1 over D=64 (cross-lane reduction in C-tile layout) ----
    float mu1[8], rs1[8];
#pragma unroll
    for (int r = 0; r < 8; ++r) {
      float s  = cvv[0][r] + cvv[1][r] + cvv[2][r] + cvv[3][r];
      float s2 = cvv[0][r] * cvv[0][r] + cvv[1][r] * cvv[1][r]
               + cvv[2][r] * cvv[2][r] + cvv[3][r] * cvv[3][r];
      s = rsum16(s); s2 = rsum16(s2);
      const float m = s * (1.f / 64.f);
      mu1[r] = m;
      rs1[r] = rsqrtf(s2 * (1.f / 64.f) - m * m + 1e-5f);
    }
    lds_fence();                       // WAR vs last iteration's ah reads
#pragma unroll
    for (int nb = 0; nb < 4; ++nb) {
      const int N = nb * 16 + nl;
#pragma unroll
      for (int r = 0; r < 8; ++r) {
        const int M = r + hf * 8;
        buf[M * 68 + N] = (cvv[nb][r] - mu1[r]) * rs1[r] * l1wv[nb] + l1bv[nb];
      }
    }
    lds_fence();
    v16h ac0 = afrag_lds(buf, 68, nl, hf, 0), ac1 = afrag_lds(buf, 68, nl, hf, 32);

    // ---- W1: D -> 2D (8 N-blocks) ----
    v8f ha[8];
#pragma unroll
    for (int nb = 0; nb < 8; ++nb) {
      v8f a = {};
      a = wmma_f16(ac0, load_bfrag(w1, 64, nb, 0, lane), a);
      a = wmma_f16(ac1, load_bfrag(w1, 64, nb, 1, lane), a);
      ha[nb] = a;
    }

    // ---- LayerNorm 2 over 2D=128 ----
    float mu2[8], rs2[8];
#pragma unroll
    for (int r = 0; r < 8; ++r) {
      float s = 0.f, s2 = 0.f;
#pragma unroll
      for (int nb = 0; nb < 8; ++nb) { s += ha[nb][r]; s2 += ha[nb][r] * ha[nb][r]; }
      s = rsum16(s); s2 = rsum16(s2);
      const float m = s * (1.f / 128.f);
      mu2[r] = m;
      rs2[r] = rsqrtf(s2 * (1.f / 128.f) - m * m + 1e-5f);
    }
    lds_fence();                       // WAR vs ac reads
#pragma unroll
    for (int nb = 0; nb < 8; ++nb) {
      const int N = nb * 16 + nl;
#pragma unroll
      for (int r = 0; r < 8; ++r) {
        const int M = r + hf * 8;
        buf[M * 132 + N] = (ha[nb][r] - mu2[r]) * rs2[r] * l2wv[nb] + l2bv[nb];
      }
    }
    lds_fence();
    v16h ah0 = afrag_lds(buf, 132, nl, hf, 0);
    v16h ah1 = afrag_lds(buf, 132, nl, hf, 32);
    v16h ah2 = afrag_lds(buf, 132, nl, hf, 64);
    v16h ah3 = afrag_lds(buf, 132, nl, hf, 96);

    // ---- W2: 2D -> D (+ bias), store mixed output (NT) ----
#pragma unroll
    for (int nb = 0; nb < 4; ++nb) {
      v8f o = {};
      o = wmma_f16(ah0, load_bfrag(w2, 128, nb, 0, lane), o);
      o = wmma_f16(ah1, load_bfrag(w2, 128, nb, 1, lane), o);
      o = wmma_f16(ah2, load_bfrag(w2, 128, nb, 2, lane), o);
      o = wmma_f16(ah3, load_bfrag(w2, 128, nb, 3, lane), o);
      const int N = nb * 16 + nl;
#pragma unroll
      for (int r = 0; r < 8; ++r) {
        const int M = r + hf * 8;
        __builtin_nontemporal_store(o[r] + b2v[nb], o1 + (size_t)M * HID_ + N);
      }
    }
  }
}

extern "C" void kernel_launch(void* const* d_in, const int* in_sizes, int n_in,
                              void* d_out, int out_size, void* d_ws, size_t ws_size,
                              hipStream_t stream) {
  (void)in_sizes; (void)n_in; (void)out_size; (void)d_ws; (void)ws_size;
  const float* gq   = (const float*)d_in[0];   // queries  [B,S,HID]
  const float* gk   = (const float*)d_in[1];   // keys
  const float* gv   = (const float*)d_in[2];   // values
  const float* gmem = (const float*)d_in[3];   // memorys  [B,H,S,D]
  const float* Wq   = (const float*)d_in[4];
  const float* Wk   = (const float*)d_in[5];
  const float* Wv   = (const float*)d_in[6];
  const float* fg   = (const float*)d_in[7];
  const float* l1w  = (const float*)d_in[8];
  const float* l1b  = (const float*)d_in[9];
  const float* W1   = (const float*)d_in[10];
  const float* l2w  = (const float*)d_in[11];
  const float* l2b  = (const float*)d_in[12];
  const float* W2   = (const float*)d_in[13];
  const float* b2   = (const float*)d_in[14];
  float* out = (float*)d_out;                  // [out | cur_mem] concatenated

  const size_t shmem = 57344 + 8 * 8448;       // 56 KB weights + 8 waves x 8.25 KB
  (void)hipFuncSetAttribute((const void*)mixer_kernel,
                            hipFuncAttributeMaxDynamicSharedMemorySize, (int)shmem);
  mixer_kernel<<<dim3(1024), dim3(256), shmem, stream>>>(
      gq, gk, gv, gmem, Wq, Wk, Wv, fg, l1w, l1b, W1, l2w, l2b, W2, b2, out);
}
